// HeteroGraph_44710609551699
// MI455X (gfx1250) — compile-verified, matrix operand order checked by reading, not run on hardware
//
#include <hip/hip_runtime.h>

// ---------------------------------------------------------------------------
// HeteroGraph GNN for MI455X (gfx1250, wave32, WMMA + TDM).
// Dense 128x128 transforms use v_wmma_f32_16x16x32_f16 with f32 accumulate;
// the f16 weight matrix is staged to LDS with the Tensor Data Mover.
// Embed+relation transforms are algebraically fused (GraphConv is linear);
// ReLU is fused into the consumers' loads (no standalone activation passes).
// ---------------------------------------------------------------------------

typedef __attribute__((ext_vector_type(16))) _Float16     v16h;
typedef __attribute__((ext_vector_type(8)))  _Float16     v8h;
typedef __attribute__((ext_vector_type(8)))  float        v8f;
typedef __attribute__((ext_vector_type(4)))  float        v4f;
typedef __attribute__((ext_vector_type(4)))  unsigned int v4u;
typedef __attribute__((ext_vector_type(4)))  int          v4i;
typedef __attribute__((ext_vector_type(8)))  int          v8i;

static inline int cdiv_i(int a, int b) { return (a + b - 1) / b; }

// ---------------------------------------------------------------------------
// C[N x 128] = relu_A?(A)[N x 128] * W[128 x 128] (+bias), W given as f16
// column-major (Wt[n*128+k] = W[k][n]).  One block = 16 rows x 128 cols;
// 8 waves, each computing a 16x16 tile via 4x K32 WMMA.  The 32KB weight
// matrix is DMA'd into LDS once per block by the Tensor Data Mover; the A
// tile is staged to LDS as f16 (with optional fused ReLU); the C tile is
// staged through LDS for b128-coalesced stores.
// N must be a multiple of 16 (all node counts here are).
// ---------------------------------------------------------------------------
template <bool HAS_BIAS, bool RELU_A>
__global__ __launch_bounds__(256) void gemm128_wmma(
    float* __restrict__ C, const float* __restrict__ A,
    const _Float16* __restrict__ Wt, const float* __restrict__ bias)
{
    __shared__ __align__(64) _Float16 As[16 * 128];    //  4 KB A tile (f16)
    __shared__ __align__(64) _Float16 Bs[128 * 128];   // 32 KB weights (f16)
    __shared__ __align__(64) float    Cs[16 * 128];    //  8 KB C tile (f32)

    const int tid = threadIdx.x;
    const int m0  = blockIdx.x * 16;

    // --- TDM: DMA the 32KB f16 weight matrix into LDS (wave 0 issues) -----
    if (tid < 32) {
        const uint32_t lds_dst = (uint32_t)(uintptr_t)(&Bs[0]);
        const uint64_t gaddr   = (uint64_t)(uintptr_t)Wt;
        v4u g0 = {};                       // D# group 0 (ISA 8.3)
        g0[0] = 1u;                                    // count=1 user D#
        g0[1] = lds_dst;                               // lds_addr
        g0[2] = (uint32_t)gaddr;                       // global_addr[31:0]
        g0[3] = (uint32_t)(gaddr >> 32) | 0x80000000u; // ga[56:32] | type=2
        v8i g1 = {};                       // D# group 1 (ISA 8.4)
        g1[0] = 3 << 16;                               // data_size = 8 bytes
        g1[1] = (int)(4096u << 16);                    // tensor_dim0 = 4096
        g1[2] = 1 << 16;                               // tensor_dim1 = 1
        g1[3] = (int)(4096u << 16);                    // tile_dim0   = 4096
        g1[5] = 4096;                                  // tensor_dim0_stride
        v4i g2 = {}, g3 = {};              // unused (<=2D tensor)
#if __clang_major__ >= 23
        v8i g1b = {};
        __builtin_amdgcn_tensor_load_to_lds(g0, g1, g2, g3, g1b, 0);
#else
        __builtin_amdgcn_tensor_load_to_lds(g0, g1, g2, g3, 0);
#endif
    }

    // --- stage the 16x128 f32 A tile into LDS as f16 (fused ReLU) ---------
    {
        const int r = tid >> 4;            // 0..15
        const int c = (tid & 15) << 3;     // 0,8,...,120
        const float* ap = A + ((size_t)(m0 + r) << 7) + c;
        v4f x0 = *(const v4f*)(ap);
        v4f x1 = *(const v4f*)(ap + 4);
        _Float16* lp = As + (r << 7) + c;
#pragma unroll
        for (int i = 0; i < 4; ++i) {
            float a0 = RELU_A ? fmaxf(x0[i], 0.0f) : x0[i];
            float a1 = RELU_A ? fmaxf(x1[i], 0.0f) : x1[i];
            lp[i]     = (_Float16)a0;
            lp[4 + i] = (_Float16)a1;
        }
    }
    if (tid < 32) __builtin_amdgcn_s_wait_tensorcnt(0);
    __syncthreads();

    const int wave = tid >> 5;             // 0..7 -> 16-col tile
    const int lane = tid & 31;
    const int n0   = wave << 4;
    const int mm   = lane & 15;            // row (A) / col (B,C)
    const int hi   = (lane >> 4) ? 8 : 0;  // K-subchunk per ISA wave32 layout

    const _Float16* wp = Bs + ((size_t)(n0 + mm) << 7);

    v8f acc = {};
#pragma unroll
    for (int ks = 0; ks < 4; ++ks) {
        const int kb = ks << 5;            // K base: 0,32,64,96
        union { v16h v; v8h h[2]; } a, b;
        // A 16x32 fragment: lane l holds row l%16; halves[0..7] = K kb+hi..,
        // halves[8..15] = K kb+16+hi..  (ISA 7.12.2, 16-bit A matrix)
        a.h[0] = *(const v8h*)(As + (mm << 7) + kb + hi);
        a.h[1] = *(const v8h*)(As + (mm << 7) + kb + 16 + hi);
        // B 32x16 fragment: lane l holds col l%16 (Bs column-major)
        b.h[0] = *(const v8h*)(wp + kb + hi);
        b.h[1] = *(const v8h*)(wp + kb + 16 + hi);
        acc = __builtin_amdgcn_wmma_f32_16x16x32_f16(
            false, a.v, false, b.v, (short)0, acc, false, false);
    }

    // --- epilogue: C/D layout row = (l/16)*8 + v, col = l%16 --------------
    const float bv    = HAS_BIAS ? bias[n0 + mm] : 0.0f;
    const int   rbase = (lane >> 4) << 3;
#pragma unroll
    for (int v = 0; v < 8; ++v)
        Cs[((rbase + v) << 7) + n0 + mm] = acc[v] + bv;
    __syncthreads();
    {
        const int r = tid >> 4;
        const int c = (tid & 15) << 3;
        const float* cp = Cs + (r << 7) + c;
        v4f y0 = *(const v4f*)cp;
        v4f y1 = *(const v4f*)(cp + 4);
        float* gp = C + ((size_t)(m0 + r) << 7) + c;
        *(v4f*)gp       = y0;
        *(v4f*)(gp + 4) = y1;
    }
}

// ---------------------------------------------------------------------------
// Y[N x 128] = X[N x K] * W[K x 128] + b   (tiny-K fused embed/message GEMM)
// ---------------------------------------------------------------------------
__global__ __launch_bounds__(256) void embed_small(
    float* __restrict__ Y, const float* __restrict__ X,
    const float* __restrict__ W, const float* __restrict__ b, int N, int K)
{
    const int gid = blockIdx.x * 256 + threadIdx.x;
    if (gid >= N * 128) return;
    const int n = gid >> 7, j = gid & 127;
    float acc = b[j];
    for (int k = 0; k < K; ++k)
        acc = fmaf(X[(size_t)n * K + k], W[(k << 7) + j], acc);
    Y[gid] = acc;
}

// acc[dst[e], :] += msg[src[e], :]   (1 thread per edge-element)
__global__ __launch_bounds__(256) void scatter_add(
    float* __restrict__ acc, const float* __restrict__ msg,
    const int* __restrict__ src, const int* __restrict__ dst, int E)
{
    const int gid = blockIdx.x * 256 + threadIdx.x;
    if (gid >= E * 128) return;
    const int e = gid >> 7, j = gid & 127;
    atomicAdd(&acc[((size_t)dst[e] << 7) + j], msg[((size_t)src[e] << 7) + j]);
}

__global__ __launch_bounds__(256) void zero_k(float* __restrict__ x, int n)
{
    const int g = blockIdx.x * 256 + threadIdx.x;
    if (g < n) x[g] = 0.0f;
}

// Wc[K x 128] = Wsmall[K x 128] @ Wrel[128 x 128]; bc = bsmall @ Wrel
__global__ __launch_bounds__(128) void combine_w(
    float* __restrict__ Wc, float* __restrict__ bc,
    const float* __restrict__ Ws, const float* __restrict__ bs,
    const float* __restrict__ Wrel, int K)
{
    const int j = threadIdx.x;
    for (int k = 0; k < K; ++k) {
        float s = 0.0f;
        for (int m = 0; m < 128; ++m)
            s = fmaf(Ws[(k << 7) + m], Wrel[(m << 7) + j], s);
        Wc[(k << 7) + j] = s;
    }
    float s = 0.0f;
    for (int m = 0; m < 128; ++m)
        s = fmaf(bs[m], Wrel[(m << 7) + j], s);
    bc[j] = s;
}

// Wt[n*128+k] = (f16)(W0+W1+W2+W3)[k*128+n]  (sum + transpose + cvt)
__global__ __launch_bounds__(256) void prep_w16(
    _Float16* __restrict__ Wt, const float* __restrict__ W0,
    const float* __restrict__ W1, const float* __restrict__ W2,
    const float* __restrict__ W3)
{
    const int gid = blockIdx.x * 256 + threadIdx.x;  // 16384 total
    const int n = gid >> 7, k = gid & 127;
    float s = W0[(k << 7) + n];
    if (W1) s += W1[(k << 7) + n];
    if (W2) s += W2[(k << 7) + n];
    if (W3) s += W3[(k << 7) + n];
    Wt[((size_t)n << 7) + k] = (_Float16)s;
}

__global__ __launch_bounds__(128) void sum_bias(
    float* __restrict__ o, const float* __restrict__ b0,
    const float* __restrict__ b1, const float* __restrict__ b2,
    const float* __restrict__ b3)
{
    const int j = threadIdx.x;
    float s = b0[j];
    if (b1) s += b1[j];
    if (b2) s += b2[j];
    if (b3) s += b3[j];
    o[j] = s;
}

// segment-sum of relu(x) into sums[G x 128] + counts (relu(op2) fused here)
__global__ __launch_bounds__(256) void pool_acc(
    float* __restrict__ sums, float* __restrict__ cnt,
    const float* __restrict__ x, const int* __restrict__ batch, int N)
{
    const int gid = blockIdx.x * 256 + threadIdx.x;
    if (gid >= N * 128) return;
    const int n = gid >> 7, j = gid & 127;
    const int g = batch[n];
    atomicAdd(&sums[((size_t)g << 7) + j], fmaxf(x[gid], 0.0f));
    if (j == 0) atomicAdd(&cnt[g], 1.0f);
}

// out[g] = dot(sums[g], Wlin) / max(cnt,1) + blin   (fused mean + linear)
__global__ __launch_bounds__(128) void pool_fin(
    float* __restrict__ out, const float* __restrict__ sums,
    const float* __restrict__ cnt, const float* __restrict__ Wlin,
    const float* __restrict__ blin)
{
    __shared__ float red[128];
    const int g = blockIdx.x, j = threadIdx.x;
    red[j] = sums[((size_t)g << 7) + j] * Wlin[j];
    __syncthreads();
    for (int s = 64; s > 0; s >>= 1) {
        if (j < s) red[j] += red[j + s];
        __syncthreads();
    }
    if (j == 0) out[g] = red[0] / fmaxf(cnt[g], 1.0f) + blin[0];
}

// ---------------------------------------------------------------------------
extern "C" void kernel_launch(void* const* d_in, const int* in_sizes, int n_in,
                              void* d_out, int out_size, void* d_ws, size_t ws_size,
                              hipStream_t stream)
{
    (void)n_in; (void)ws_size;
    const float* x_op   = (const float*)d_in[0];
    const float* x_tab  = (const float*)d_in[1];
    const float* x_col  = (const float*)d_in[2];
    const float* x_pred = (const float*)d_in[3];
    const int* scan_src = (const int*)d_in[4];
    const int* scan_dst = (const int*)d_in[5];
    const int* filt_src = (const int*)d_in[6];
    const int* filt_dst = (const int*)d_in[7];
    const int* out_src  = (const int*)d_in[8];
    const int* out_dst  = (const int*)d_in[9];
    const int* conn_src = (const int*)d_in[10];
    const int* conn_dst = (const int*)d_in[11];
    const int* call_src = (const int*)d_in[12];
    const int* call_dst = (const int*)d_in[13];
    const int* batch    = (const int*)d_in[14];

    int i = 15;
    const float* W_op   = (const float*)d_in[i++]; const float* b_op   = (const float*)d_in[i++];
    const float* W_tab  = (const float*)d_in[i++]; const float* b_tab  = (const float*)d_in[i++];
    const float* W_col  = (const float*)d_in[i++]; const float* b_col  = (const float*)d_in[i++];
    const float* W_pred = (const float*)d_in[i++]; const float* b_pred = (const float*)d_in[i++];
    struct GCp { const float *Wrel, *brel, *Wroot; };
    GCp gc[7];  // c1_scan, c1_filt, c1_out, c1_conn, c1_call, c2_filt, c2_call
    for (int c = 0; c < 7; ++c) {
        gc[c].Wrel  = (const float*)d_in[i++];
        gc[c].brel  = (const float*)d_in[i++];
        gc[c].Wroot = (const float*)d_in[i++];
    }
    const GCp& c1_scan = gc[0]; const GCp& c1_filt = gc[1]; const GCp& c1_out = gc[2];
    const GCp& c1_conn = gc[3]; const GCp& c1_call = gc[4];
    const GCp& c2_filt = gc[5]; const GCp& c2_call = gc[6];
    const float* W_lin = (const float*)d_in[i++];
    const float* b_lin = (const float*)d_in[i++];

    const int N_OP   = in_sizes[0] / 4;
    const int N_TAB  = in_sizes[1] / 2;
    const int N_COL  = in_sizes[2] / 16;
    const int N_PRED = in_sizes[3];
    const int E_SCAN = in_sizes[4], E_FILT = in_sizes[6], E_OUT = in_sizes[8];
    const int E_CONN = in_sizes[10], E_CALL = in_sizes[12];
    const int G = out_size;
    float* out = (float*)d_out;

    // Workspace layout (f32 elements); ~412 MB total.
    float* F = (float*)d_ws;
    const size_t SZ_OP = (size_t)N_OP * 128, SZ_PRED = (size_t)N_PRED * 128;
    float* opA   = F;                       // op_h   -> (conv2) op accumulator (pre-relu op2)
    float* opB   = opA + SZ_OP;             // conv1 op accumulator (pre-relu op1)
    float* opC   = opB + SZ_OP;             // message scratch (<= 200K rows)
    float* predA = opC + SZ_OP;             // pred_h -> (conv2) pred messages
    float* predB = predA + SZ_PRED;         // conv1 pred accumulator (pre-relu pred1)
    float* sums  = predB + SZ_PRED;         // G x 128
    float* cnt   = sums + (size_t)G * 128;  // G
    float* bsum  = cnt + G;                 // 128
    float* Wc    = bsum + 128;              // 16 x 128 combined weight
    float* bc    = Wc + 16 * 128;           // 128
    _Float16* Wt = (_Float16*)(bc + 128);   // 128 x 128 f16 (8192 f32 slots)

    const float* NF = nullptr;

#define EMB(Y, X, W, B, N, K) \
    embed_small<<<cdiv_i((N) * 128, 256), 256, 0, stream>>>(Y, X, W, B, N, K)
#define SCAT(ACC, MSG, S, D, E) \
    scatter_add<<<cdiv_i((E) * 128, 256), 256, 0, stream>>>(ACC, MSG, S, D, E)
#define GEMM(BIASF, RELUF, Cp, Ap, BIAS, N) \
    gemm128_wmma<BIASF, RELUF><<<(N) / 16, 256, 0, stream>>>(Cp, Ap, Wt, BIAS)
#define PREPW(W0, W1, W2, W3) \
    prep_w16<<<64, 256, 0, stream>>>(Wt, W0, W1, W2, W3)
#define COMBW(WS, BS, WREL, K) \
    combine_w<<<1, 128, 0, stream>>>(Wc, bc, WS, BS, WREL, K)

    // ---- raw embeddings needed beyond a single relation ----
    EMB(opA,   x_op,   W_op,   b_op,   N_OP,   4);   // op_h
    EMB(predA, x_pred, W_pred, b_pred, N_PRED, 1);   // pred_h

    // ---- conv1, operator dst: root term + summed relation biases ----
    PREPW(c1_scan.Wroot, c1_filt.Wroot, c1_out.Wroot, c1_call.Wroot);
    sum_bias<<<1, 128, 0, stream>>>(bsum, c1_scan.brel, c1_filt.brel,
                                    c1_out.brel, c1_call.brel);
    GEMM(true, false, opB, opA, bsum, N_OP);         // opB = op_h @ SumWroot + Sum b_rel

    // scan: table -> operator   (fused: x_tab @ (W_tab @ Wrel))
    COMBW(W_tab, b_tab, c1_scan.Wrel, 2);
    EMB(opC, x_tab, Wc, bc, N_TAB, 2);
    SCAT(opB, opC, scan_src, scan_dst, E_SCAN);

    // filters: predicate -> operator
    COMBW(W_pred, b_pred, c1_filt.Wrel, 1);
    EMB(opC, x_pred, Wc, bc, N_PRED, 1);
    SCAT(opB, opC, filt_src, filt_dst, E_FILT);

    // outputs: column -> operator
    COMBW(W_col, b_col, c1_out.Wrel, 16);
    EMB(opC, x_col, Wc, bc, N_COL, 16);
    SCAT(opB, opC, out_src, out_dst, E_OUT);

    // calledby: operator -> operator (full 128x128 WMMA transform)
    PREPW(c1_call.Wrel, NF, NF, NF);
    GEMM(false, false, opC, opA, NF, N_OP);
    SCAT(opB, opC, call_src, call_dst, E_CALL);
    // opB now holds pre-relu op1; relu fused into its consumers below.

    // ---- conv1, predicate dst ----
    PREPW(c1_conn.Wroot, NF, NF, NF);
    GEMM(true, false, predB, predA, c1_conn.brel, N_PRED);
    COMBW(W_col, b_col, c1_conn.Wrel, 16);
    EMB(opC, x_col, Wc, bc, N_COL, 16);
    SCAT(predB, opC, conn_src, conn_dst, E_CONN);
    // predB now holds pre-relu pred1; relu fused into its consumer below.

    // ---- conv2 (operator dst only): filters + calledby ----
    PREPW(c2_filt.Wroot, c2_call.Wroot, NF, NF);
    sum_bias<<<1, 128, 0, stream>>>(bsum, c2_filt.brel, c2_call.brel, NF, NF);
    GEMM(true, true, opA, opB, bsum, N_OP);          // relu(op1) @ SumWroot + Sum b_rel

    PREPW(c2_filt.Wrel, NF, NF, NF);
    GEMM(false, true, predA, predB, NF, N_PRED);     // relu(pred1) @ Wrel_c2filt
    SCAT(opA, predA, filt_src, filt_dst, E_FILT);

    PREPW(c2_call.Wrel, NF, NF, NF);
    GEMM(false, true, opC, opB, NF, N_OP);           // relu(op1) @ Wrel_c2call
    SCAT(opA, opC, call_src, call_dst, E_CALL);
    // opA now holds pre-relu op2; relu fused into pool_acc.

    // ---- global mean pool + final linear ----
    zero_k<<<cdiv_i(G * 128 + G, 256), 256, 0, stream>>>(sums, G * 128 + G);
    pool_acc<<<cdiv_i(N_OP * 128, 256), 256, 0, stream>>>(sums, cnt, opA, batch, N_OP);
    pool_fin<<<G, 128, 0, stream>>>(out, sums, cnt, W_lin, b_lin);

#undef EMB
#undef SCAT
#undef GEMM
#undef PREPW
#undef COMBW
}
